// VQ_78323023610348
// MI455X (gfx1250) — matrix operand order repaired and checked by reading, not compiled, not run on hardware
//
#include <hip/hip_runtime.h>
#include <stdint.h>

typedef __attribute__((ext_vector_type(16))) __bf16 v16bf;
typedef __attribute__((ext_vector_type(8)))  float  v8f;

union BF16x16 { v16bf v; uint4 q; unsigned short u[16]; };

__device__ __forceinline__ unsigned short f2bf(float f) {
  unsigned int u = __float_as_uint(f);
  u += 0x7FFFu + ((u >> 16) & 1u);          // round-to-nearest-even
  return (unsigned short)(u >> 16);
}
__device__ __forceinline__ float bf2f(unsigned short b) {
  return __uint_as_float(((unsigned int)b) << 16);
}
__device__ __forceinline__ float h2f(unsigned short b) {
  union { unsigned short u; _Float16 h; } c; c.u = b; return (float)c.h;
}

#define SROW 1032   // f16 stride per row (1024 + 8 pad -> conflict-free banks)
#define ROWS 64     // rows per workgroup
// s is staged pre-scaled by log2(e)/8 so pass 2 can use raw v_exp_f32 (exp2).
#define NEG_L2E_O8 (-0.18033688011112042f)   // -0.125 * log2(e)

// ---------------- prep: codebook -> bf16 hi/lo split + row norms ----------------
__global__ __launch_bounds__(256) void vq_prep_kernel(
    const float* __restrict__ cb, unsigned short* __restrict__ cb_hi,
    unsigned short* __restrict__ cb_lo, float* __restrict__ c2) {
  const int tid  = threadIdx.x;
  const int code = blockIdx.x * 16 + (tid >> 4);
  const int part = tid & 15;
  float4 v = ((const float4*)cb)[code * 16 + part];
  float f[4] = {v.x, v.y, v.z, v.w};
  unsigned short h[4], l[4];
  float ss = 0.0f;
#pragma unroll
  for (int i = 0; i < 4; ++i) {
    h[i] = f2bf(f[i]);
    l[i] = f2bf(f[i] - bf2f(h[i]));
    ss += f[i] * f[i];
  }
  uint2 ph, pl;
  ph.x = (unsigned)h[0] | ((unsigned)h[1] << 16);
  ph.y = (unsigned)h[2] | ((unsigned)h[3] << 16);
  pl.x = (unsigned)l[0] | ((unsigned)l[1] << 16);
  pl.y = (unsigned)l[2] | ((unsigned)l[3] << 16);
  ((uint2*)cb_hi)[code * 16 + part] = ph;
  ((uint2*)cb_lo)[code * 16 + part] = pl;
#pragma unroll
  for (int m = 8; m; m >>= 1) ss += __shfl_xor(ss, m, 32);
  if (part == 0) c2[code] = ss;
}

// ---------------- main: WMMA distances + softmax + noisy argmax + gather ----------------
extern __shared__ char vq_smem[];

__global__ __launch_bounds__(256) void vq_main_kernel(
    const float* __restrict__ x, const float* __restrict__ noise,
    const unsigned short* __restrict__ cb_hi, const unsigned short* __restrict__ cb_lo,
    const float* __restrict__ c2, const float* __restrict__ codebook,
    const int* __restrict__ noise_level, float* __restrict__ out) {
  _Float16* s_s   = (_Float16*)vq_smem;                     // [64][1032] f16
  float*    s_x2  = (float*)(vq_smem + ROWS * SROW * 2);    // [64]
  int*      s_idx = (int*)(s_x2 + ROWS);                    // [64]

  const int  tid    = threadIdx.x;
  const int  wave   = tid >> 5;
  const int  lane   = tid & 31;
  const long wg_row = (long)blockIdx.x * ROWS;

  // ---- prefetch this WG's noise block (256 KB) so the HBM stream overlaps the GEMM ----
  {
    const char* nbase = (const char*)(noise + wg_row * 1024);
#pragma unroll
    for (int i = 0; i < 8; ++i)
      __builtin_prefetch(nbase + ((size_t)tid + 256 * i) * 128, 0, 1);
  }

  // ---- pass 0: exact f32 row norms ||x||^2 (wave handles 8 rows) ----
#pragma unroll
  for (int r8 = 0; r8 < 8; ++r8) {
    int row = wave * 8 + r8;
    const float* xr = x + (wg_row + row) * 64;
    float a = xr[lane], b = xr[lane + 32];
    float ss = a * a + b * b;
#pragma unroll
    for (int m = 16; m; m >>= 1) ss += __shfl_xor(ss, m, 32);
    if (lane == 0) s_x2[row] = ss;
  }

  // ---- A fragments: split bf16 (x = hi + lo), two K=32 chunks ----
  // 16-bit A layout: lane<16 -> row=lane, K in {0..7, 16..23}; lane>=16 -> K in {8..15, 24..31}
  const int rt   = wave & 3;   // row tile 0..3
  const int ch   = wave >> 2;  // column half 0..1
  const int mloc = lane & 15;
  const int hi   = lane >> 4;
  BF16x16 a_hi[2], a_lo[2];
  {
    const float4* xr4 = (const float4*)(x + (wg_row + rt * 16 + mloc) * 64);
#pragma unroll
    for (int kk = 0; kk < 2; ++kk) {
      float4 p[4];
      p[0] = xr4[kk * 8 + hi * 2 + 0];
      p[1] = xr4[kk * 8 + hi * 2 + 1];
      p[2] = xr4[kk * 8 + hi * 2 + 4];
      p[3] = xr4[kk * 8 + hi * 2 + 5];
#pragma unroll
      for (int c = 0; c < 4; ++c) {
        float f[4] = {p[c].x, p[c].y, p[c].z, p[c].w};
#pragma unroll
        for (int i = 0; i < 4; ++i) {
          unsigned short hb = f2bf(f[i]);
          a_hi[kk].u[c * 4 + i] = hb;
          a_lo[kk].u[c * 4 + i] = f2bf(f[i] - bf2f(hb));
        }
      }
    }
  }
  __syncthreads();

  float x2v[8];
#pragma unroll
  for (int v = 0; v < 8; ++v) x2v[v] = s_x2[rt * 16 + hi * 8 + v];

  // ---- pass 1: d^2 = x2 + c2 - 2*dot via split-bf16 WMMA; stage s2 = -d*log2e/8 to LDS ----
  // B layout (K x 16): lane holds column j = lane%16, K half selected by lane/16 ->
  // 8 bf16 of codebook row j at k0 = 8*(lane/16) + 32*kk, then 8 more at k0 + 16 (same order as A).
  const uint4* cbhq = (const uint4*)cb_hi;
  const uint4* cblq = (const uint4*)cb_lo;
#pragma unroll 2
  for (int t = 0; t < 32; ++t) {
    int jt = ch * 32 + t;
    int j  = jt * 16 + mloc;
    float c2v = c2[j];
    int bb = j * 8 + hi;                           // uint4 units into codebook row
    BF16x16 bh0, bh1, bl0, bl1;
    bh0.q = cbhq[bb];     bh1.q = cbhq[bb + 4];
    ((uint4*)&bh0)[1] = cbhq[bb + 2];     ((uint4*)&bh1)[1] = cbhq[bb + 6];
    bl0.q = cblq[bb];     bl1.q = cblq[bb + 4];
    ((uint4*)&bl0)[1] = cblq[bb + 2];     ((uint4*)&bl1)[1] = cblq[bb + 6];
    // two independent accumulator chains (K-chunk 0 / K-chunk 1), interleaved
    v8f acc0 = {}, acc1 = {};
    acc0 = __builtin_amdgcn_wmma_f32_16x16x32_bf16(false, a_lo[0].v, false, bh0.v, (short)0, acc0, false, false);
    acc1 = __builtin_amdgcn_wmma_f32_16x16x32_bf16(false, a_lo[1].v, false, bh1.v, (short)0, acc1, false, false);
    acc0 = __builtin_amdgcn_wmma_f32_16x16x32_bf16(false, a_hi[0].v, false, bl0.v, (short)0, acc0, false, false);
    acc1 = __builtin_amdgcn_wmma_f32_16x16x32_bf16(false, a_hi[1].v, false, bl1.v, (short)0, acc1, false, false);
    acc0 = __builtin_amdgcn_wmma_f32_16x16x32_bf16(false, a_hi[0].v, false, bh0.v, (short)0, acc0, false, false);
    acc1 = __builtin_amdgcn_wmma_f32_16x16x32_bf16(false, a_hi[1].v, false, bh1.v, (short)0, acc1, false, false);
#pragma unroll
    for (int v = 0; v < 8; ++v) {                  // C layout: lane<16 -> M=v, lane>=16 -> M=v+8
      float dot = acc0[v] + acc1[v];
      float d2  = x2v[v] + c2v - 2.0f * dot;
      float s   = NEG_L2E_O8 * __builtin_amdgcn_sqrtf(fmaxf(d2, 0.0f));
      s_s[(rt * 16 + hi * 8 + v) * SROW + j] = (_Float16)s;
    }
  }
  __syncthreads();

  // ---- pass 2: per-row max, Z, then argmax(exp2(s-m) - Z*nl*noise) fused with noise stream ----
  const float lvl = (float)noise_level[0];
#pragma unroll 1
  for (int r8 = 0; r8 < 8; ++r8) {
    int row = wave * 8 + r8;
    const unsigned* sr32 = (const unsigned*)(s_s + row * SROW);
    float m = -3.0e38f;
    for (int it = 0; it < 16; ++it) {
      unsigned w = sr32[lane + 32 * it];
      m = fmaxf(m, h2f((unsigned short)w));
      m = fmaxf(m, h2f((unsigned short)(w >> 16)));
    }
#pragma unroll
    for (int k = 16; k; k >>= 1) m = fmaxf(m, __shfl_xor(m, k, 32));
    float z = 0.0f;
    for (int it = 0; it < 16; ++it) {
      unsigned w = sr32[lane + 32 * it];
      z += __builtin_amdgcn_exp2f(h2f((unsigned short)w) - m);
      z += __builtin_amdgcn_exp2f(h2f((unsigned short)(w >> 16)) - m);
    }
#pragma unroll
    for (int k = 16; k; k >>= 1) z += __shfl_xor(z, k, 32);
    float zl = z * lvl;   // argmax(r - nl*noise) == argmax(exp2(s-m) - Z*nl*noise), Z>0

    const float4* nr4  = (const float4*)(noise + (wg_row + row) * 1024);
    const uint2*  sr64 = (const uint2*)sr32;
    float best = -3.0e38f; int bidx = 0;
    for (int it = 0; it < 8; ++it) {
      int q = lane + 32 * it;
      float4 nv = nr4[q];
      uint2  sw = sr64[q];
      int j0 = q * 4;
      float e0 = __builtin_amdgcn_exp2f(h2f((unsigned short)sw.x)         - m) - zl * nv.x;
      float e1 = __builtin_amdgcn_exp2f(h2f((unsigned short)(sw.x >> 16)) - m) - zl * nv.y;
      float e2 = __builtin_amdgcn_exp2f(h2f((unsigned short)sw.y)         - m) - zl * nv.z;
      float e3 = __builtin_amdgcn_exp2f(h2f((unsigned short)(sw.y >> 16)) - m) - zl * nv.w;
      if (e0 > best) { best = e0; bidx = j0;     }
      if (e1 > best) { best = e1; bidx = j0 + 1; }
      if (e2 > best) { best = e2; bidx = j0 + 2; }
      if (e3 > best) { best = e3; bidx = j0 + 3; }
    }
#pragma unroll
    for (int k = 16; k; k >>= 1) {   // first-occurrence tie break like jnp.argmax
      float ob = __shfl_xor(best, k, 32);
      int   oi = __shfl_xor(bidx, k, 32);
      if (ob > best || (ob == best && oi < bidx)) { best = ob; bidx = oi; }
    }
    if (lane == 0) s_idx[row] = bidx;
  }
  __syncthreads();

  // ---- output: y[row] = codebook[idx[row]] (coalesced float4 gather) ----
  const float4* cb4  = (const float4*)codebook;
  float4*       out4 = (float4*)out;
#pragma unroll
  for (int k = 0; k < 4; ++k) {
    int flat = tid + 256 * k;        // 0..1023 = 64 rows x 16 float4
    int row  = flat >> 4;
    int comp = flat & 15;
    int code = s_idx[row];
    out4[(wg_row + row) * 16 + comp] = cb4[code * 16 + comp];
  }
}

extern "C" void kernel_launch(void* const* d_in, const int* in_sizes, int n_in,
                              void* d_out, int out_size, void* d_ws, size_t ws_size,
                              hipStream_t stream) {
  const float* x        = (const float*)d_in[0];
  const float* codebook = (const float*)d_in[1];
  const float* noise    = (const float*)d_in[2];
  const int*   nl       = (const int*)d_in[3];
  float* out = (float*)d_out;

  const int N  = in_sizes[0] / 64;   // 131072 rows
  const int NC = in_sizes[1] / 64;   // 1024 codes

  unsigned short* cb_hi = (unsigned short*)d_ws;                 // 128 KB
  unsigned short* cb_lo = cb_hi + (size_t)NC * 64;               // 128 KB
  float*          c2    = (float*)(cb_lo + (size_t)NC * 64);     // 4 KB

  vq_prep_kernel<<<NC / 16, 256, 0, stream>>>(codebook, cb_hi, cb_lo, c2);

  const size_t smem = (size_t)ROWS * SROW * 2 + (size_t)ROWS * 4 + (size_t)ROWS * 4; // ~129.5 KB
  hipFuncSetAttribute((const void*)vq_main_kernel,
                      hipFuncAttributeMaxDynamicSharedMemorySize, (int)smem);
  vq_main_kernel<<<N / ROWS, 256, smem, stream>>>(x, noise, cb_hi, cb_lo, c2,
                                                  codebook, nl, out);
}
